// CrossAttention_16595753632018
// MI455X (gfx1250) — matrix operand order repaired and checked by reading, not compiled, not run on hardware
//
#include <hip/hip_runtime.h>
#include <hip/hip_bf16.h>
#include <math.h>

#define DIMD 1024
#define NROWS 8192
#define MROWS 8192

typedef __attribute__((ext_vector_type(16))) _Float16 v16h;
typedef __attribute__((ext_vector_type(8)))  float    v8f;

union FragU {          // one 16x32 f16 A-frag or 32x16 f16 B-frag (8 VGPRs)
    uint4    u4[2];
    _Float16 h[16];
    v16h     v;
};
union Half8 {          // 8 packed f16 = 16 bytes
    uint4    u;
    _Float16 h[8];
};

// ---------------------------------------------------------------------------
// Q/K/V projection GEMM:  Out[r,c] = (f16)( A[r,:] @ W[:,c] + bias[c] )
// A: [rows x 1024] f32, W: [1024 x 1024] f32, bias: [1024] f32
// TRANSPOSED=true stores Out[c*ROWS + r] (produces V^T for the attention).
// Block tile 64x128, BK=32, 8 waves as 2x4; wave tile 32x32 = 4 WMMAs with
// 2 A-frag + 2 B-frag reuse per k-step.
// ---------------------------------------------------------------------------
template <bool TRANSPOSED>
__global__ __launch_bounds__(256)
void qkv_gemm(const float* __restrict__ A, const float* __restrict__ W,
              const float* __restrict__ bias, _Float16* __restrict__ out)
{
    constexpr int BM = 64, BN = 128, BK = 32;
    constexpr int ROWS = NROWS;
    __shared__ __align__(16) _Float16 As[BM * BK];   // [m][k], stride 32
    __shared__ __align__(16) _Float16 Bs[BN * BK];   // [n][k] (transposed), stride 32

    const int tid  = threadIdx.x;
    const int wave = tid >> 5;
    const int lane = tid & 31;
    const int hl   = lane >> 4;
    const int ln   = lane & 15;
    const int wm   = wave >> 2;     // 0..1  -> row offset wm*32
    const int wn   = wave & 3;      // 0..3  -> col offset wn*32
    const int m0   = blockIdx.x * BM;
    const int n0   = blockIdx.y * BN;

    // cooperative-load indices
    const int la_m = tid >> 2;          // 0..63
    const int la_k = (tid & 3) * 8;     // 0,8,16,24
    const int lb_n = tid & 127;         // 0..127
    const int lb_k = (tid >> 7) * 8;    // 0 or 8

    v8f acc[2][2] = {};

    for (int k0 = 0; k0 < DIMD; k0 += BK) {
        __syncthreads();
        // ---- A tile: 64x32 f32 -> f16 LDS (row-major) ----
        {
            const float* src = A + (size_t)(m0 + la_m) * DIMD + k0 + la_k;
            float4 f0 = *(const float4*)(src);
            float4 f1 = *(const float4*)(src + 4);
            Half8 hp;
            hp.h[0] = (_Float16)f0.x; hp.h[1] = (_Float16)f0.y;
            hp.h[2] = (_Float16)f0.z; hp.h[3] = (_Float16)f0.w;
            hp.h[4] = (_Float16)f1.x; hp.h[5] = (_Float16)f1.y;
            hp.h[6] = (_Float16)f1.z; hp.h[7] = (_Float16)f1.w;
            *(uint4*)&As[la_m * BK + la_k] = hp.u;
        }
        // ---- B tile: 32x128 f32 -> transposed f16 LDS [n][k] ----
        #pragma unroll
        for (int half = 0; half < 2; ++half) {
            const int kb = lb_k + half * 16;     // 0,8,16,24 over the two halves
            const float* src = W + (size_t)(k0 + kb) * DIMD + n0 + lb_n;
            Half8 hp;
            #pragma unroll
            for (int i = 0; i < 8; ++i) hp.h[i] = (_Float16)src[(size_t)i * DIMD];
            *(uint4*)&Bs[lb_n * BK + kb] = hp.u;
        }
        __syncthreads();

        // ---- fragments + 4 WMMAs with reuse ----
        FragU a[2], b[2];
        #pragma unroll
        for (int i = 0; i < 2; ++i) {
            const _Float16* ap = &As[(wm * 32 + i * 16 + ln) * BK];
            a[i].u4[0] = *(const uint4*)(ap + 8 * hl);
            a[i].u4[1] = *(const uint4*)(ap + 16 + 8 * hl);
        }
        #pragma unroll
        for (int j = 0; j < 2; ++j) {
            const _Float16* bp = &Bs[(wn * 32 + j * 16 + ln) * BK];
            b[j].u4[0] = *(const uint4*)(bp + 8 * hl);
            b[j].u4[1] = *(const uint4*)(bp + 16 + 8 * hl);
        }
        #pragma unroll
        for (int i = 0; i < 2; ++i)
            #pragma unroll
            for (int j = 0; j < 2; ++j)
                acc[i][j] = __builtin_amdgcn_wmma_f32_16x16x32_f16(
                    false, a[i].v, false, b[j].v, (short)0, acc[i][j], false, false);
    }

    // ---- epilogue: + bias, convert f16, store (compile-time layout) ----
    #pragma unroll
    for (int j = 0; j < 2; ++j) {
        const int gcol = n0 + wn * 32 + j * 16 + ln;
        const float bvv = bias[gcol];
        #pragma unroll
        for (int r = 0; r < 8; ++r) {
            int row0 = m0 + wm * 32 + r + 8 * hl;
            int row1 = row0 + 16;
            _Float16 v0 = (_Float16)(acc[0][j][r] + bvv);
            _Float16 v1 = (_Float16)(acc[1][j][r] + bvv);
            if constexpr (!TRANSPOSED) {
                out[(size_t)row0 * DIMD + gcol] = v0;
                out[(size_t)row1 * DIMD + gcol] = v1;
            } else {
                out[(size_t)gcol * ROWS + row0] = v0;
                out[(size_t)gcol * ROWS + row1] = v1;
            }
        }
    }
}

// ---------------------------------------------------------------------------
// Fused flash attention over f16 Q [N,D], K [M,D], V^T [D,M]; f32 output.
// Block = 256 threads (8 waves), 32 query rows per block.
// Wave w owns output-dim slice [w*128, w*128+128).
// Online softmax stats (m, l, alpha) + padded 32x32 score tile in LDS.
// ---------------------------------------------------------------------------
#define SR_STRIDE 33   // float stride: 33*i+c distinct mod 64 -> conflict-free
#define PB_STRIDE 40   // f16 stride: 80B rows, 16B aligned, conflict-free b128

__global__ __launch_bounds__(256)
void flash_attn(const _Float16* __restrict__ Q, const _Float16* __restrict__ K,
                const _Float16* __restrict__ Vt, float* __restrict__ out)
{
    __shared__ float                  Sred[32 * SR_STRIDE];
    __shared__ __align__(16) _Float16 Pb[32 * PB_STRIDE];
    __shared__ float mrow[32], lrow[32], arow[32];

    const int tid  = threadIdx.x;
    const int wave = tid >> 5;
    const int lane = tid & 31;
    const int hl   = lane >> 4;
    const int ln   = lane & 15;
    const int r0   = blockIdx.x * 32;   // query-row base
    const int d0   = wave * 128;        // dim slice base

    // ---- load Q fragments once (pre-scaled by 1/sqrt(D) = 1/32) ----
    FragU Qf[2][4];
    #pragma unroll
    for (int tm = 0; tm < 2; ++tm) {
        #pragma unroll
        for (int kt = 0; kt < 4; ++kt) {
            const _Float16* qp = Q + (size_t)(r0 + tm * 16 + ln) * DIMD + d0 + kt * 32;
            Qf[tm][kt].u4[0] = *(const uint4*)(qp + 8 * hl);
            Qf[tm][kt].u4[1] = *(const uint4*)(qp + 16 + 8 * hl);
            #pragma unroll
            for (int e = 0; e < 16; ++e)
                Qf[tm][kt].h[e] = Qf[tm][kt].h[e] * (_Float16)0.03125f;
        }
    }

    v8f O[2][8] = {};   // 32 x 128 f32 output accumulators per wave

    if (tid < 32) { mrow[tid] = -INFINITY; lrow[tid] = 0.0f; }

    for (int j = 0; j < MROWS; j += 32) {
        __syncthreads();
        for (int i = tid; i < 32 * SR_STRIDE; i += 256) Sred[i] = 0.0f;
        __syncthreads();

        // prefetch next key tile (L2-resident; hides latency behind WMMA)
        if (j + 32 < MROWS) {
            __builtin_prefetch(K  + (size_t)(j + 32 + lane % 32) * DIMD + d0, 0, 0);
            __builtin_prefetch(Vt + (size_t)(d0 + lane % 32) * MROWS + j + 32, 0, 0);
        }

        // ---- partial S = Q_slice @ K_slice^T over this wave's 128 dims ----
        v8f S[2][2] = {};
        #pragma unroll
        for (int kt = 0; kt < 4; ++kt) {
            FragU Kf[2];
            #pragma unroll
            for (int tn = 0; tn < 2; ++tn) {
                const _Float16* kp = K + (size_t)(j + tn * 16 + ln) * DIMD
                                       + d0 + kt * 32;
                Kf[tn].u4[0] = *(const uint4*)(kp + 8 * hl);
                Kf[tn].u4[1] = *(const uint4*)(kp + 16 + 8 * hl);
            }
            #pragma unroll
            for (int tm = 0; tm < 2; ++tm)
                #pragma unroll
                for (int tn = 0; tn < 2; ++tn)
                    S[tm][tn] = __builtin_amdgcn_wmma_f32_16x16x32_f16(
                        false, Qf[tm][kt].v, false, Kf[tn].v,
                        (short)0, S[tm][tn], false, false);
        }
        // ---- cross-wave reduction of S into LDS (conflict-free stride) ----
        #pragma unroll
        for (int tm = 0; tm < 2; ++tm)
            #pragma unroll
            for (int tn = 0; tn < 2; ++tn)
                #pragma unroll
                for (int r = 0; r < 8; ++r) {
                    int row = tm * 16 + r + 8 * hl;
                    int col = tn * 16 + ln;
                    atomicAdd(&Sred[row * SR_STRIDE + col], S[tm][tn][r]);
                }
        __syncthreads();

        // ---- online softmax: one lane per query row (wave 0) ----
        if (tid < 32) {
            const int i = tid;
            float mold = mrow[i];
            float mx = mold;
            for (int c = 0; c < 32; ++c) mx = fmaxf(mx, Sred[i * SR_STRIDE + c]);
            float alpha = __expf(mold - mx);
            float s = 0.0f;
            for (int c = 0; c < 32; ++c) {
                float p = __expf(Sred[i * SR_STRIDE + c] - mx);
                s += p;
                Pb[i * PB_STRIDE + c] = (_Float16)p;
            }
            mrow[i] = mx;
            lrow[i] = lrow[i] * alpha + s;
            arow[i] = alpha;
        }
        __syncthreads();

        // ---- rescale O, then O += P @ V_slice ----
        float al[2][8];
        #pragma unroll
        for (int tm = 0; tm < 2; ++tm)
            #pragma unroll
            for (int r = 0; r < 8; ++r)
                al[tm][r] = arow[tm * 16 + r + 8 * hl];

        FragU Pf[2];
        #pragma unroll
        for (int tm = 0; tm < 2; ++tm) {
            const _Float16* pp = &Pb[(tm * 16 + ln) * PB_STRIDE];
            Pf[tm].u4[0] = *(const uint4*)(pp + 8 * hl);
            Pf[tm].u4[1] = *(const uint4*)(pp + 16 + 8 * hl);
        }
        #pragma unroll
        for (int nt = 0; nt < 8; ++nt) {
            FragU Vf;
            const _Float16* vp = Vt + (size_t)(d0 + nt * 16 + ln) * MROWS + j;
            Vf.u4[0] = *(const uint4*)(vp + 8 * hl);
            Vf.u4[1] = *(const uint4*)(vp + 16 + 8 * hl);
            #pragma unroll
            for (int tm = 0; tm < 2; ++tm) {
                v8f o = O[tm][nt];
                #pragma unroll
                for (int r = 0; r < 8; ++r) o[r] = o[r] * al[tm][r];
                O[tm][nt] = __builtin_amdgcn_wmma_f32_16x16x32_f16(
                    false, Pf[tm].v, false, Vf.v, (short)0, o, false, false);
            }
        }
    }
    __syncthreads();

    // ---- normalize by l and store f32 output ----
    float li[2][8];
    #pragma unroll
    for (int tm = 0; tm < 2; ++tm)
        #pragma unroll
        for (int r = 0; r < 8; ++r)
            li[tm][r] = 1.0f / lrow[tm * 16 + r + 8 * hl];

    #pragma unroll
    for (int tm = 0; tm < 2; ++tm)
        #pragma unroll
        for (int nt = 0; nt < 8; ++nt)
            #pragma unroll
            for (int r = 0; r < 8; ++r) {
                int row = r0 + tm * 16 + r + 8 * hl;
                int col = d0 + nt * 16 + ln;
                out[(size_t)row * DIMD + col] = O[tm][nt][r] * li[tm][r];
            }
}

// ---------------------------------------------------------------------------
extern "C" void kernel_launch(void* const* d_in, const int* in_sizes, int n_in,
                              void* d_out, int out_size, void* d_ws, size_t ws_size,
                              hipStream_t stream)
{
    const float* x       = (const float*)d_in[0];   // [8192,1024]
    const float* context = (const float*)d_in[1];   // [8192,1024]
    const float* Wq      = (const float*)d_in[2];
    const float* bq      = (const float*)d_in[3];
    const float* Wk      = (const float*)d_in[4];
    const float* bk      = (const float*)d_in[5];
    const float* Wv      = (const float*)d_in[6];
    const float* bv      = (const float*)d_in[7];
    float* outp          = (float*)d_out;

    const size_t bytesQ = (size_t)NROWS * DIMD * sizeof(_Float16);  // 16 MB
    char* ws = (char*)d_ws;
    _Float16* Qh = (_Float16*)(ws);
    _Float16* Kh = (_Float16*)(ws + bytesQ);
    _Float16* Vt = (_Float16*)(ws + 2 * bytesQ);

    dim3 ggrid(NROWS / 64, DIMD / 128);
    qkv_gemm<false><<<ggrid, 256, 0, stream>>>(x,       Wq, bq, Qh);
    qkv_gemm<false><<<ggrid, 256, 0, stream>>>(context, Wk, bk, Kh);
    qkv_gemm<true ><<<ggrid, 256, 0, stream>>>(context, Wv, bv, Vt);

    flash_attn<<<NROWS / 32, 256, 0, stream>>>(Qh, Kh, Vt, outp);
}
